// Context_ComplEx_47399259078994
// MI455X (gfx1250) — matrix operand order repaired and checked by reading, not compiled, not run on hardware
//
#include <hip/hip_runtime.h>
#include <hip/hip_bf16.h>

#define N_ENT   100000
#define N_REL   1000
#define RANK    256
#define BATCH   1024
#define MAX_NB  50
#define TWO_R   512
#define ROWS    8

#define SCORES_OFF 102400000L   // BATCH * N_ENT

typedef __attribute__((ext_vector_type(16))) __bf16 bf16x16;
typedef __attribute__((ext_vector_type(8)))  float  f32x8;

// ---------------------------------------------------------------------------
// Kernel 1: convert E_ent fp32 -> bf16 copy (102.4 MB, L2-resident afterwards)
// ---------------------------------------------------------------------------
__global__ void convert_bf16_kernel(const float4* __restrict__ src,
                                    ushort4* __restrict__ dst, long n4) {
  long i = (long)blockIdx.x * blockDim.x + threadIdx.x;
  long stride = (long)gridDim.x * blockDim.x;
  for (; i < n4; i += stride) {
    float4 f = src[i];
    union { __bf16 h; unsigned short u; } c0, c1, c2, c3;
    c0.h = (__bf16)f.x; c1.h = (__bf16)f.y;
    c2.h = (__bf16)f.z; c3.h = (__bf16)f.w;
    dst[i] = make_ushort4(c0.u, c1.u, c2.u, c3.u);
  }
}

// ---------------------------------------------------------------------------
// Kernel 2: n_lhs, n_rel, n_rhs  (elementwise, gathered)
// ---------------------------------------------------------------------------
__global__ __launch_bounds__(256) void norms_kernel(const int* __restrict__ x,
                                                    const float* __restrict__ E_ent,
                                                    const float* __restrict__ E_rel,
                                                    float* __restrict__ out) {
  int b = blockIdx.x, k = threadIdx.x;
  int i0 = x[b * 3 + 0], i1 = x[b * 3 + 1], i2 = x[b * 3 + 2];
  float l0 = E_ent[(long)i0 * TWO_R + k],       l1 = E_ent[(long)i0 * TWO_R + RANK + k];
  float r0 = E_rel[(long)i1 * TWO_R + k],       r1 = E_rel[(long)i1 * TWO_R + RANK + k];
  float o0 = E_ent[(long)i2 * TWO_R + k],       o1 = E_ent[(long)i2 * TWO_R + RANK + k];
  out[SCORES_OFF +          (long)b * RANK + k] = sqrtf(l0 * l0 + l1 * l1);
  out[SCORES_OFF + 262144 + (long)b * RANK + k] = sqrtf(r0 * r0 + r1 * r1);
  out[SCORES_OFF + 524288 + (long)b * RANK + k] = sqrtf(o0 * o0 + o1 * o1);
}

// ---------------------------------------------------------------------------
// Kernel 3: w0/w1 = complex GEMM (B,512)x(512,256); 8 rows/block to amortize W
// ---------------------------------------------------------------------------
__global__ __launch_bounds__(256) void w_kernel(const int* __restrict__ x,
                                                const float* __restrict__ E_ent,
                                                const float* __restrict__ E_rel,
                                                const float* __restrict__ W0,
                                                const float* __restrict__ W1,
                                                const float* __restrict__ bw0,
                                                const float* __restrict__ bw1,
                                                float* __restrict__ w0o,
                                                float* __restrict__ w1o) {
  __shared__ float t0s[ROWS][TWO_R];
  __shared__ float t1s[ROWS][TWO_R];
  int tid = threadIdx.x;
  int b0 = blockIdx.x * ROWS;
  for (int r = 0; r < ROWS; ++r) {
    int b = b0 + r;
    int i0 = x[b * 3 + 0], i1 = x[b * 3 + 1];
    for (int k = tid; k < TWO_R; k += 256) {
      float a0, a1;
      if (k < RANK) {
        a0 = E_ent[(long)i0 * TWO_R + k];
        a1 = E_ent[(long)i0 * TWO_R + RANK + k];
      } else {
        a0 = E_rel[(long)i1 * TWO_R + (k - RANK)];
        a1 = E_rel[(long)i1 * TWO_R + RANK + (k - RANK)];
      }
      t0s[r][k] = a0; t1s[r][k] = a1;
    }
  }
  __syncthreads();
  int j = tid;
  float acc0[ROWS], acc1[ROWS];
  for (int r = 0; r < ROWS; ++r) { acc0[r] = 0.f; acc1[r] = 0.f; }
  for (int k = 0; k < TWO_R; ++k) {
    float a = W0[k * RANK + j];
    float c = W1[k * RANK + j];
#pragma unroll
    for (int r = 0; r < ROWS; ++r) {
      float u = t0s[r][k], v = t1s[r][k];
      acc0[r] += u * a - v * c;
      acc1[r] += u * c + v * a;
    }
  }
  float bb0 = bw0[j], bb1 = bw1[j];
  for (int r = 0; r < ROWS; ++r) {
    w0o[(long)(b0 + r) * RANK + j] = acc0[r] + bb0;
    w1o[(long)(b0 + r) * RANK + j] = acc1[r] + bb1;
  }
}

// ---------------------------------------------------------------------------
// Kernel 4: neighbor attention -> ec0, ec1
// ---------------------------------------------------------------------------
__global__ __launch_bounds__(256) void attn_kernel(const int* __restrict__ nb_idx,
                                                   const float* __restrict__ E_ent,
                                                   const float* __restrict__ w0i,
                                                   const float* __restrict__ w1i,
                                                   float* __restrict__ ec0o,
                                                   float* __restrict__ ec1o) {
  __shared__ float w0s[RANK], w1s[RANK];
  __shared__ int   nbs[MAX_NB];
  __shared__ float lg[MAX_NB];
  __shared__ float alpha[MAX_NB];
  int b = blockIdx.x, tid = threadIdx.x;
  w0s[tid] = w0i[(long)b * RANK + tid];
  w1s[tid] = w1i[(long)b * RANK + tid];
  if (tid < MAX_NB) nbs[tid] = nb_idx[b * MAX_NB + tid];
  __syncthreads();
  int wv = tid >> 5, lane = tid & 31;
  for (int m = wv; m < MAX_NB; m += 8) {
    long base = (long)nbs[m] * TWO_R;
    float s = 0.f;
#pragma unroll
    for (int ii = 0; ii < 8; ++ii) {
      int k = lane + ii * 32;
      s += w0s[k] * E_ent[base + k] - w1s[k] * E_ent[base + RANK + k];
    }
    for (int off = 16; off > 0; off >>= 1) s += __shfl_xor(s, off, 32);
    if (lane == 0) lg[m] = s;
  }
  __syncthreads();
  if (tid == 0) {
    float mx = lg[0];
    for (int m = 1; m < MAX_NB; ++m) mx = fmaxf(mx, lg[m]);
    float sum = 0.f;
    for (int m = 0; m < MAX_NB; ++m) { float e = __expf(lg[m] - mx); alpha[m] = e; sum += e; }
    float inv = 1.f / sum;
    for (int m = 0; m < MAX_NB; ++m) alpha[m] *= inv;
  }
  __syncthreads();
  int k = tid;
  float a0 = 0.f, a1 = 0.f;
  for (int m = 0; m < MAX_NB; ++m) {
    long base = (long)nbs[m] * TWO_R;
    float al = alpha[m];
    a0 += al * E_ent[base + k];
    a1 += al * E_ent[base + RANK + k];
  }
  ec0o[(long)b * RANK + k] = a0;
  ec1o[(long)b * RANK + k] = a1;
}

// ---------------------------------------------------------------------------
// Kernel 5: ec@W2, gate, q_re/q_im -> bf16 Q, n_gec
// ---------------------------------------------------------------------------
__global__ __launch_bounds__(256) void post_kernel(const int* __restrict__ x,
                                                   const float* __restrict__ E_ent,
                                                   const float* __restrict__ E_rel,
                                                   const float* __restrict__ ec0i,
                                                   const float* __restrict__ ec1i,
                                                   const float* __restrict__ W20,
                                                   const float* __restrict__ W21,
                                                   const float* __restrict__ bw20,
                                                   const float* __restrict__ bw21,
                                                   const float* __restrict__ Uo0,
                                                   const float* __restrict__ Uo1,
                                                   const float* __restrict__ Wo0,
                                                   const float* __restrict__ b_g,
                                                   unsigned short* __restrict__ Qbf,
                                                   float* __restrict__ out) {
  __shared__ float e0s[ROWS][RANK];
  __shared__ float e1s[ROWS][RANK];
  __shared__ float red[8];
  __shared__ float gsh;
  int tid = threadIdx.x;
  int b0 = blockIdx.x * ROWS;
  for (int r = 0; r < ROWS; ++r) {
    e0s[r][tid] = ec0i[(long)(b0 + r) * RANK + tid];
    e1s[r][tid] = ec1i[(long)(b0 + r) * RANK + tid];
  }
  __syncthreads();
  int j = tid;
  float n0[ROWS], n1[ROWS];
  for (int r = 0; r < ROWS; ++r) { n0[r] = 0.f; n1[r] = 0.f; }
  for (int k = 0; k < RANK; ++k) {
    float a = W20[k * RANK + j], c = W21[k * RANK + j];
#pragma unroll
    for (int r = 0; r < ROWS; ++r) {
      float u = e0s[r][k], v = e1s[r][k];
      n0[r] += u * a - v * c;
      n1[r] += u * c + v * a;
    }
  }
  float bb0 = bw20[j], bb1 = bw21[j];
  float u0 = Uo0[j], u1 = Uo1[j], wo = Wo0[j], bg = b_g[0];
  int wv = tid >> 5, lane = tid & 31;
  for (int r = 0; r < ROWS; ++r) {
    int b = b0 + r;
    int i0 = x[b * 3 + 0], i1 = x[b * 3 + 1];
    float l0 = E_ent[(long)i0 * TWO_R + j], l1 = E_ent[(long)i0 * TWO_R + RANK + j];
    float r0v = E_rel[(long)i1 * TWO_R + j], r1v = E_rel[(long)i1 * TWO_R + RANK + j];
    float ecn0 = n0[r] + bb0, ecn1 = n1[r] + bb1;
    float srrr = l0 * r0v, siri = l1 * r1v, sirr = l1 * r0v, srri = l0 * r1v;
    float part = (srrr - siri) * u0 - (sirr + srri) * u1 + ecn0 * wo;
    for (int off = 16; off > 0; off >>= 1) part += __shfl_xor(part, off, 32);
    if (lane == 0) red[wv] = part;
    __syncthreads();
    if (tid == 0) {
      float s = 0.f;
      for (int w2 = 0; w2 < 8; ++w2) s += red[w2];
      gsh = 1.f / (1.f + __expf(-(s + bg)));
    }
    __syncthreads();
    float g = gsh;
    float gec0 = g * ecn0 + (1.f - g);
    float gec1 = g * ecn1;
    float q_re = (srrr - siri) * gec0 + (sirr + srri) * gec1;
    float q_im = (srri + sirr) * gec0 + (siri - srrr) * gec1;
    union { __bf16 h; unsigned short u; } cv;
    cv.h = (__bf16)q_re; Qbf[(long)b * TWO_R + j] = cv.u;
    cv.h = (__bf16)q_im; Qbf[(long)b * TWO_R + RANK + j] = cv.u;
    out[SCORES_OFF + 786432 + (long)b * RANK + j] = sqrtf(gec0 * gec0 + gec1 * gec1);
    __syncthreads();   // protect red/gsh before next row
  }
}

// ---------------------------------------------------------------------------
// Kernel 6: scores = Q(bf16) @ Ebf16^T via v_wmma_f32_16x16x32_bf16
//   workgroup = 8 waves = 128M x 128N tile; B tile staged through LDS
// ---------------------------------------------------------------------------
#define NT  128
#define KT  32
#define LDB 40   // padded LDS row stride (bf16 elems) to spread banks

__global__ __launch_bounds__(256) void score_gemm_kernel(const unsigned short* __restrict__ Qbf,
                                                         const unsigned short* __restrict__ Ebf,
                                                         float* __restrict__ out) {
  __shared__ alignas(16) unsigned short Bt[NT * LDB];   // 10,240 B
  int tid  = threadIdx.x;
  int wv   = tid >> 5, lane = tid & 31;
  int hi   = lane >> 4;       // which 16-lane half
  int l16  = lane & 15;
  long Nbase = (long)blockIdx.x * NT;
  int  Mbase = blockIdx.y * 128 + wv * 16;

  f32x8 acc[8];
#pragma unroll
  for (int j = 0; j < 8; ++j)
#pragma unroll
    for (int i = 0; i < 8; ++i) acc[j][i] = 0.f;

  const unsigned short* qrow = Qbf + (long)(Mbase + l16) * TWO_R;

  union AB { uint4 u[2]; bf16x16 v; };

  for (int kk = 0; kk < TWO_R / KT; ++kk) {
    int K0 = kk * KT;
    // ---- stage B tile: 128 rows x 32 K bf16 -> LDS ----
#pragma unroll
    for (int c = tid; c < NT * 4; c += 256) {
      int row = c >> 2, cc = c & 3;
      uint4 v = make_uint4(0u, 0u, 0u, 0u);
      long gr = Nbase + row;
      if (gr < N_ENT) v = *(const uint4*)(Ebf + gr * TWO_R + K0 + cc * 8);
      *(uint4*)(&Bt[row * LDB + cc * 8]) = v;
    }
    __syncthreads();

    // prefetch next K-slice of this N tile (global_prefetch_b8)
    if (kk + 1 < TWO_R / KT) {
      long gr = Nbase + (tid >> 1);
      if (gr < N_ENT)
        __builtin_prefetch(Ebf + gr * TWO_R + (kk + 1) * KT + (tid & 1) * 16, 0, 0);
    }

    // ---- A fragment: row (Mbase+l16); lane<16 K{0..7,16..23}, lane>=16 K{8..15,24..31}
    AB a;
    a.u[0] = *(const uint4*)(qrow + K0 + hi * 8);
    a.u[1] = *(const uint4*)(qrow + K0 + 16 + hi * 8);

#pragma unroll
    for (int j = 0; j < 8; ++j) {
      // B fragment: column (j*16+l16); lane<16 K{0..7,16..23}, lane>=16 K{8..15,24..31}
      AB bm;
      const unsigned short* bp = &Bt[(j * 16 + l16) * LDB + hi * 8];
      bm.u[0] = *(const uint4*)(bp);
      bm.u[1] = *(const uint4*)(bp + 16);
      acc[j] = __builtin_amdgcn_wmma_f32_16x16x32_bf16(
          false, a.v, false, bm.v, (short)0, acc[j], false, false);
    }
    __syncthreads();
  }

  // ---- store D: lanes 0-15 -> M=i, lanes 16-31 -> M=8+i; N = Nbase+16j+l16
#pragma unroll
  for (int j = 0; j < 8; ++j) {
    long n = Nbase + j * 16 + l16;
    if (n < N_ENT) {
#pragma unroll
      for (int i = 0; i < 8; ++i) {
        int m = Mbase + i + hi * 8;
        out[(long)m * N_ENT + n] = acc[j][i];
      }
    }
  }
}

// ---------------------------------------------------------------------------
extern "C" void kernel_launch(void* const* d_in, const int* in_sizes, int n_in,
                              void* d_out, int out_size, void* d_ws, size_t ws_size,
                              hipStream_t stream) {
  const int*   x     = (const int*)d_in[0];
  const int*   nb    = (const int*)d_in[1];
  const float* E_ent = (const float*)d_in[2];
  const float* E_rel = (const float*)d_in[3];
  const float* W0    = (const float*)d_in[4];
  const float* W1    = (const float*)d_in[5];
  const float* bw0   = (const float*)d_in[6];
  const float* bw1   = (const float*)d_in[7];
  const float* W20   = (const float*)d_in[8];
  const float* W21   = (const float*)d_in[9];
  const float* bw20  = (const float*)d_in[10];
  const float* bw21  = (const float*)d_in[11];
  const float* Uo0   = (const float*)d_in[12];
  const float* Uo1   = (const float*)d_in[13];
  const float* Wo0   = (const float*)d_in[14];
  const float* b_g   = (const float*)d_in[15];
  float* out = (float*)d_out;
  char*  ws  = (char*)d_ws;

  const size_t E_BYTES = (size_t)N_ENT * TWO_R * 2;   // 102,400,000
  const size_t Q_BYTES = (size_t)BATCH * TWO_R * 2;   //   1,048,576
  unsigned short* Ebf = (unsigned short*)ws;
  unsigned short* Qbf = (unsigned short*)(ws + E_BYTES);
  float* w0  = (float*)(ws + E_BYTES + Q_BYTES);
  float* w1  = w0  + (size_t)BATCH * RANK;
  float* ec0 = w1  + (size_t)BATCH * RANK;
  float* ec1 = ec0 + (size_t)BATCH * RANK;

  long n4 = (long)N_ENT * TWO_R / 4;
  convert_bf16_kernel<<<50000, 256, 0, stream>>>((const float4*)E_ent, (ushort4*)Ebf, n4);
  norms_kernel<<<BATCH, 256, 0, stream>>>(x, E_ent, E_rel, out);
  w_kernel<<<BATCH / ROWS, 256, 0, stream>>>(x, E_ent, E_rel, W0, W1, bw0, bw1, w0, w1);
  attn_kernel<<<BATCH, 256, 0, stream>>>(nb, E_ent, w0, w1, ec0, ec1);
  post_kernel<<<BATCH / ROWS, 256, 0, stream>>>(x, E_ent, E_rel, ec0, ec1, W20, W21,
                                                bw20, bw21, Uo0, Uo1, Wo0, b_g, Qbf, out);
  dim3 g((N_ENT + NT - 1) / NT, BATCH / 128);
  score_gemm_kernel<<<g, 256, 0, stream>>>(Qbf, Ebf, out);
}
